// Wavlm_spk_Moe_2980707304236
// MI455X (gfx1250) — matrix operand order repaired and checked by reading, not compiled, not run on hardware
//
#include <hip/hip_runtime.h>
#include <hip/hip_bf16.h>
#include <math.h>
#include <stdint.h>

// ---------------------------------------------------------------------------
// MoE of 8 bidirectional 2-layer LSTMs (B=32,T=1024,D=H=128) on gfx1250.
// bf16 WMMA (v_wmma_f32_16x16x32_bf16) everywhere; recurrence fused with the
// input projection; all 16 (expert,dir) chains run concurrently.
// Input tiles staged per-timestep with the Tensor Data Mover
// (tensor_load_to_lds + s_wait_tensorcnt) when the toolchain exposes it.
// Workspace requirement: ~300 MB.
// ---------------------------------------------------------------------------

typedef __bf16 bf16_t;
typedef __attribute__((ext_vector_type(8)))  bf16_t bf16x8;
typedef __attribute__((ext_vector_type(16))) bf16_t v16bf;
typedef __attribute__((ext_vector_type(8)))  float  v8f;
typedef __attribute__((ext_vector_type(4)))  unsigned int u32x4;
typedef __attribute__((ext_vector_type(4)))  int i32x4;
typedef __attribute__((ext_vector_type(8)))  int i32x8;

#ifndef __has_builtin
#define __has_builtin(x) 0
#endif
#if __has_builtin(__builtin_amdgcn_tensor_load_to_lds) && __has_builtin(__builtin_amdgcn_s_wait_tensorcnt)
#define HAVE_TDM 1
#else
#define HAVE_TDM 0
#endif

#define B_  32
#define T_  1024
#define D_  128
#define H_  128
#define E_  8
#define O_  44
#define S_  256
#define G4_ 512   // 4H
#define F2_ 256   // 2H

__device__ __forceinline__ bf16_t f2bf(float f) {
  union { float f; unsigned u; } a; a.f = f;
  unsigned r = (a.u + 0x7FFFu + ((a.u >> 16) & 1u)) >> 16;
  union { unsigned short s; bf16_t b; } o; o.s = (unsigned short)r;
  return o.b;
}
__device__ __forceinline__ float bf2f(bf16_t b) {
  union { unsigned short s; bf16_t b; } i; i.b = b;
  union { unsigned u; float f; } o; o.u = ((unsigned)i.s) << 16;
  return o.f;
}
__device__ __forceinline__ float sigmoidf_(float x) { return 1.0f / (1.0f + __expf(-x)); }

__device__ __forceinline__ v8f wmma_bf16(v16bf a, v16bf b, v8f c) {
  return __builtin_amdgcn_wmma_f32_16x16x32_bf16(false, a, false, b, (short)0, c, false, false);
}

// A fragment (16x32 bf16). base points at (row0, k0=0); row_stride in elements.
// lane<16 holds K k0+0..7 and k0+16..23 ; lane>=16 holds K k0+8..15 and k0+24..31.
__device__ __forceinline__ v16bf load_a(const bf16_t* base, size_t row_stride, int k0, int lane) {
  const int l15 = lane & 15, lhi = lane >> 4;
  const bf16_t* p = base + (size_t)l15 * row_stride + k0 + lhi * 8;
  union { v16bf v; bf16x8 h[2]; } u;
  u.h[0] = *(const bf16x8*)(p);
  u.h[1] = *(const bf16x8*)(p + 16);
  return u.v;
}

// B fragment (32x16 bf16) from row-major weight W[N,K] (computing X @ W^T).
__device__ __forceinline__ v16bf load_b(const bf16_t* W, int ldk, int n0, int k0, int lane) {
  const int l15 = lane & 15, lhi = lane >> 4;
  const bf16_t* p = W + (size_t)(n0 + l15) * ldk + k0 + lhi * 8;
  union { v16bf v; bf16x8 h[2]; } u;
  u.h[0] = *(const bf16x8*)(p);
  u.h[1] = *(const bf16x8*)(p + 16);
  return u.v;
}

#if HAVE_TDM
// Low 32 bits of a generic LDS pointer == LDS byte offset (flat->LDS truncation).
__device__ __forceinline__ unsigned lds_off(const void* p) {
  return (unsigned)(unsigned long long)(uintptr_t)p;
}
// Issue a TDM 2D tile load: 32 rows x Fin bf16 elements, row stride T_*Fin elems,
// from global (b=0, t, k=0) into LDS at lds_byte (packed [32][Fin]).
__device__ __forceinline__ void tdm_load_x(const bf16_t* base, int t, int Fin, unsigned lds_byte) {
  const bf16_t* gsrc = base + (size_t)t * Fin;
  unsigned long long ga = (unsigned long long)(uintptr_t)gsrc;
  const unsigned dim0 = (unsigned)Fin;     // tensor/tile width (elements)
  const unsigned dim1 = 32u;               // rows
  const unsigned long long st0 = (unsigned long long)T_ * (unsigned)Fin;  // row stride (elements)
  u32x4 g0;
  g0[0] = 1u;                                              // count=1, user descriptor
  g0[1] = lds_byte;                                        // lds_addr
  g0[2] = (unsigned)ga;                                    // global_addr[31:0]
  g0[3] = (unsigned)((ga >> 32) & 0x01FFFFFFull) | 0x80000000u;  // addr[56:32] | type=2
  i32x8 g1;
  g1[0] = 1 << 16;                                         // workgroup_mask=0, data_size=2B
  g1[1] = (int)((dim0 & 0xFFFFu) << 16);                   // tensor_dim0[15:0]
  g1[2] = (int)(((dim0 >> 16) & 0xFFFFu) | ((dim1 & 0xFFFFu) << 16));   // dim0 hi | dim1 lo
  g1[3] = (int)(((dim1 >> 16) & 0xFFFFu) | ((dim0 & 0xFFFFu) << 16));   // dim1 hi | tile_dim0
  g1[4] = (int)dim1;                                       // tile_dim1=32, tile_dim2=0
  g1[5] = (int)(st0 & 0xFFFFFFFFull);                      // tensor_dim0_stride lo
  g1[6] = (int)((st0 >> 32) & 0xFFFFull);                  // stride hi | dim1_stride lo = 0
  g1[7] = 0;
  i32x4 z4 = {0, 0, 0, 0};
#if __clang_major__ >= 23
  i32x8 z8 = {0, 0, 0, 0, 0, 0, 0, 0};
  __builtin_amdgcn_tensor_load_to_lds(g0, g1, z4, z4, z8, 0);
#else
  __builtin_amdgcn_tensor_load_to_lds(g0, g1, z4, z4, 0);
#endif
}
#endif  // HAVE_TDM

// ------------------------------- small utils -------------------------------

__global__ void cvt_bf16_kernel(const float* __restrict__ src, bf16_t* __restrict__ dst, int n) {
  int i = blockIdx.x * 256 + threadIdx.x;
  if (i < n) dst[i] = f2bf(src[i]);
}

__global__ void bias_sum_kernel(const float* __restrict__ a, const float* __restrict__ b,
                                float* __restrict__ o, int n) {
  int i = blockIdx.x * 256 + threadIdx.x;
  if (i < n) o[i] = a[i] + b[i];
}

// pad fc_w [44,128] -> bf16 [48,128] (zero rows 44..47)
__global__ void fcw_pad_kernel(const float* __restrict__ src, bf16_t* __restrict__ dst) {
  int i = blockIdx.x * 256 + threadIdx.x;
  if (i < 48 * H_) {
    int r = i >> 7, c = i & 127;
    dst[i] = f2bf(r < O_ ? src[r * H_ + c] : 0.f);
  }
}

// ----------------------------- gating + loss -------------------------------

__device__ __forceinline__ float cv_sq(const float* v) {
  float m = 0.f;
  for (int i = 0; i < 8; ++i) m += v[i];
  m *= 0.125f;
  float var = 0.f;
  for (int i = 0; i < 8; ++i) { float d = v[i] - m; var += d * d; }
  var *= (1.f / 7.f);                 // ddof=1
  return var / (m * m + 1e-10f);
}

__global__ void gate_loss_kernel(const float* __restrict__ spk, const float* __restrict__ wgw,
                                 const float* __restrict__ wgb, float* __restrict__ gates,
                                 float* __restrict__ loss_out) {
  __shared__ float sg[B_ * E_];
  const int b = threadIdx.x;   // 32 threads
  float lg[E_];
  for (int e = 0; e < E_; ++e) {
    float s = wgb[e];
    const float* sp = spk + b * S_;
    const float* wp = wgw + e * S_;
    for (int k = 0; k < S_; ++k) s += sp[k] * wp[k];
    lg[e] = s;
  }
  float tmp[E_];
  for (int e = 0; e < E_; ++e) tmp[e] = lg[e];
  int   idx[4]; float tv[4];
  for (int k = 0; k < 4; ++k) {
    int bi = 0; float bv = -1e30f;
    for (int e = 0; e < E_; ++e) if (tmp[e] > bv) { bv = tmp[e]; bi = e; }
    idx[k] = bi; tv[k] = bv; tmp[bi] = -1e30f;
  }
  float den = 0.f, ex[4];
  for (int k = 0; k < 4; ++k) { ex[k] = __expf(tv[k] - tv[0]); den += ex[k]; }
  float g[E_];
  for (int e = 0; e < E_; ++e) g[e] = 0.f;
  for (int k = 0; k < 4; ++k) g[idx[k]] = ex[k] / den;
  for (int e = 0; e < E_; ++e) { gates[b * E_ + e] = g[e]; sg[b * E_ + e] = g[e]; }
  __syncthreads();
  if (b == 0) {
    float imp[E_], ld[E_];
    for (int e = 0; e < E_; ++e) {
      float si = 0.f, sl = 0.f;
      for (int bb = 0; bb < B_; ++bb) { float v = sg[bb * E_ + e]; si += v; sl += (v > 0.f) ? 1.f : 0.f; }
      imp[e] = si; ld[e] = sl;
    }
    loss_out[0] = (cv_sq(imp) + cv_sq(ld)) * 0.01f;
  }
}

// ------------------------- fused LSTM scan (one layer) ---------------------
// grid.x = E_*2 : one workgroup per (expert, direction), 8 waves.
// Per timestep: g[32,512] = x_t @ Wih^T + h @ Whh^T + bias, fully via WMMA.
// h state: LDS bf16; c state: 16 regs/lane; gates exchanged via LDS bf16.
// x_t tile staged into LDS by the Tensor Data Mover (double-buffered for
// Fin=128; single-buffered for Fin=256).

__global__ __launch_bounds__(256) void lstm_scan_kernel(
    const bf16_t* __restrict__ in, size_t in_expert_stride, int Fin,
    const bf16_t* __restrict__ Wih, const bf16_t* __restrict__ Whh,
    const float* __restrict__ biasc, bf16_t* __restrict__ hout) {
  const int eid = blockIdx.x >> 1;
  const int dir = blockIdx.x & 1;
  const int tid = threadIdx.x;
  const int w = tid >> 5, lane = tid & 31;
  const int l15 = lane & 15, lhi = lane >> 4;

  __shared__ bf16_t hbuf[B_ * H_];        // 8 KB: h_{t-1}, bf16
  __shared__ bf16_t gact[B_ * G4_];       // 32 KB: activated gates
#if HAVE_TDM
  __shared__ bf16_t xstage[2 * B_ * 128]; // 16 KB: TDM-staged x_t tile(s)
#endif

  const bf16_t* inE  = in + (size_t)eid * in_expert_stride;
  const bf16_t* WihD = Wih + (size_t)(eid * 2 + dir) * G4_ * Fin;
  const bf16_t* WhhD = Whh + (size_t)(eid * 2 + dir) * G4_ * H_;
  const float*  bsD  = biasc + (size_t)(eid * 2 + dir) * G4_;
  bf16_t* houtD = hout + (size_t)eid * B_ * T_ * F2_ + (size_t)dir * H_;

  for (int i = tid; i < B_ * H_; i += 256) hbuf[i] = f2bf(0.f);
  float c[16];
#pragma unroll
  for (int j = 0; j < 16; ++j) c[j] = 0.f;
  const int m_c = tid >> 3;               // batch row owned for c/h update
  const int hcb = (tid & 7) * 16;         // h-column base owned

  float biasv[4];
#pragma unroll
  for (int n = 0; n < 4; ++n) biasv[n] = bsD[w * 64 + n * 16 + l15];

#if HAVE_TDM
  const bool db = (Fin == 128);           // double-buffer when tile is 8 KB
  const unsigned xbase = lds_off(&xstage[0]);
  if (db && w == 0) {                     // stage first tile
    tdm_load_x(inE, dir ? (T_ - 1) : 0, Fin, xbase);
    __builtin_amdgcn_s_wait_tensorcnt(0);
  }
#endif
  __syncthreads();

  for (int tt = 0; tt < T_; ++tt) {
    const int t = dir ? (T_ - 1 - tt) : tt;
    v8f acc[2][4] = {};

#if HAVE_TDM
    const int cur = db ? (tt & 1) : 0;
    if (!db) {                            // single-buffered: load, wait, publish
      if (w == 0) {
        tdm_load_x(inE, t, Fin, xbase);
        __builtin_amdgcn_s_wait_tensorcnt(0);
      }
      __syncthreads();
    } else if (w == 0 && tt + 1 < T_) {   // prefetch next tile into other half
      tdm_load_x(inE, dir ? (t - 1) : (t + 1), Fin, xbase + (unsigned)(1 - cur) * (B_ * 128 * 2));
    }
    const bf16_t* xb = &xstage[cur * (B_ * 128)];
    // ---- input projection: x_t @ Wih^T (A from LDS stage) ----
    for (int k0 = 0; k0 < Fin; k0 += 32) {
      v16bf a0 = load_a(xb, Fin, k0, lane);                 // b 0..15
      v16bf a1 = load_a(xb + 16 * Fin, Fin, k0, lane);      // b 16..31
#pragma unroll
      for (int n = 0; n < 4; ++n) {
        v16bf bb = load_b(WihD, Fin, w * 64 + n * 16, k0, lane);
        acc[0][n] = wmma_bf16(a0, bb, acc[0][n]);
        acc[1][n] = wmma_bf16(a1, bb, acc[1][n]);
      }
    }
#else
    // fallback: A fragments straight from global + software prefetch
    if (tt + 1 < T_) {
      const int tn = dir ? (T_ - 2 - tt) : (tt + 1);
      __builtin_prefetch(inE + ((size_t)lane * T_ + tn) * Fin, 0, 1);
    }
    for (int k0 = 0; k0 < Fin; k0 += 32) {
      v16bf a0 = load_a(inE + (size_t)t * Fin, (size_t)T_ * Fin, k0, lane);
      v16bf a1 = load_a(inE + ((size_t)16 * T_ + t) * Fin, (size_t)T_ * Fin, k0, lane);
#pragma unroll
      for (int n = 0; n < 4; ++n) {
        v16bf bb = load_b(WihD, Fin, w * 64 + n * 16, k0, lane);
        acc[0][n] = wmma_bf16(a0, bb, acc[0][n]);
        acc[1][n] = wmma_bf16(a1, bb, acc[1][n]);
      }
    }
#endif
    // ---- recurrent: h_{t-1} @ Whh^T  (K = 128, A from LDS) ----
#pragma unroll
    for (int k0 = 0; k0 < H_; k0 += 32) {
      v16bf a0 = load_a(&hbuf[0], H_, k0, lane);
      v16bf a1 = load_a(&hbuf[16 * H_], H_, k0, lane);
#pragma unroll
      for (int n = 0; n < 4; ++n) {
        v16bf bb = load_b(WhhD, H_, w * 64 + n * 16, k0, lane);
        acc[0][n] = wmma_bf16(a0, bb, acc[0][n]);
        acc[1][n] = wmma_bf16(a1, bb, acc[1][n]);
      }
    }
    // ---- bias + gate activations -> LDS ----
#pragma unroll
    for (int mt = 0; mt < 2; ++mt) {
#pragma unroll
      for (int n = 0; n < 4; ++n) {
        const int nt = w * 4 + n;
        const int gate = nt >> 3;          // 0=i 1=f 2=g 3=o
        const int col = nt * 16 + l15;
#pragma unroll
        for (int v = 0; v < 8; ++v) {
          const int m = mt * 16 + lhi * 8 + v;
          float val = acc[mt][n][v] + biasv[n];
          float r = (gate == 2) ? tanhf(val) : sigmoidf_(val);
          gact[m * G4_ + col] = f2bf(r);
        }
      }
    }
    __syncthreads();
    // ---- c/h update (each lane owns 16 (m,hc) cells) ----
#pragma unroll
    for (int j = 0; j < 16; ++j) {
      const int hc = hcb + j;
      float iv = bf2f(gact[m_c * G4_ + hc]);
      float fv = bf2f(gact[m_c * G4_ + 128 + hc]);
      float gv = bf2f(gact[m_c * G4_ + 256 + hc]);
      float ov = bf2f(gact[m_c * G4_ + 384 + hc]);
      c[j] = fv * c[j] + iv * gv;
      float h = ov * tanhf(c[j]);
      bf16_t hb = f2bf(h);
      hbuf[m_c * H_ + hc] = hb;
      houtD[((size_t)m_c * T_ + t) * F2_ + hc] = hb;
    }
#if HAVE_TDM
    if (db && w == 0 && tt + 1 < T_) __builtin_amdgcn_s_wait_tensorcnt(0);
#endif
    __syncthreads();
  }
}

// ---------- fc1 + gate-weighted mixture over experts (no atomics) ----------
// dbf[b,t,:] = sum_e gates[b,e] * (h1[e,b,t,:] @ fc1_w^T) + (sum_e g)*fc1_b

__global__ __launch_bounds__(256) void fc1_gate_kernel(
    const bf16_t* __restrict__ h1, const bf16_t* __restrict__ fc1wb,
    const float* __restrict__ fc1b, const float* __restrict__ gates,
    bf16_t* __restrict__ dbf) {
  const int w = threadIdx.x >> 5, lane = threadIdx.x & 31;
  const int l15 = lane & 15, lhi = lane >> 4;
  const int row0 = blockIdx.x * 128 + w * 16;   // M = B*T
  const int n0 = blockIdx.y * 64;               // N = 128
  const int b = row0 >> 10;                     // T = 1024 divides tile rows
  v8f res[4] = {};
  float gsum = 0.f;
  for (int e = 0; e < E_; ++e) {
    const float gt = gates[b * E_ + e];
    gsum += gt;
    if (gt == 0.f) continue;
    v8f acc[4] = {};
    const bf16_t* A = h1 + ((size_t)e * B_ * T_ + row0) * F2_;
    for (int k0 = 0; k0 < F2_; k0 += 32) {
      v16bf a = load_a(A, F2_, k0, lane);
#pragma unroll
      for (int j = 0; j < 4; ++j) {
        v16bf bb = load_b(fc1wb, F2_, n0 + j * 16, k0, lane);
        acc[j] = wmma_bf16(a, bb, acc[j]);
      }
    }
#pragma unroll
    for (int j = 0; j < 4; ++j)
#pragma unroll
      for (int v = 0; v < 8; ++v) res[j][v] += gt * acc[j][v];
  }
#pragma unroll
  for (int j = 0; j < 4; ++j) {
    const int col = n0 + j * 16 + l15;
    const float bias = gsum * fc1b[col];
#pragma unroll
    for (int v = 0; v < 8; ++v) {
      const int row = row0 + lhi * 8 + v;
      dbf[(size_t)row * H_ + col] = f2bf(res[j][v] + bias);
    }
  }
}

// ----------------- final fc: out[t,b,o] = d @ fc_w^T + fc_b ----------------

__global__ __launch_bounds__(256) void fc_out_kernel(
    const bf16_t* __restrict__ dbf, const bf16_t* __restrict__ fcwb,
    const float* __restrict__ fcb, float* __restrict__ out) {
  const int w = threadIdx.x >> 5, lane = threadIdx.x & 31;
  const int l15 = lane & 15, lhi = lane >> 4;
  const int row0 = blockIdx.x * 128 + w * 16;
  v8f acc[3] = {};
  for (int k0 = 0; k0 < H_; k0 += 32) {
    v16bf a = load_a(dbf + (size_t)row0 * H_, H_, k0, lane);
#pragma unroll
    for (int j = 0; j < 3; ++j) {
      v16bf bb = load_b(fcwb, H_, j * 16, k0, lane);
      acc[j] = wmma_bf16(a, bb, acc[j]);
    }
  }
#pragma unroll
  for (int j = 0; j < 3; ++j) {
    const int o = j * 16 + l15;
    if (o < O_) {
      const float bias = fcb[o];
#pragma unroll
      for (int v = 0; v < 8; ++v) {
        const int row = row0 + lhi * 8 + v;
        const int b = row >> 10, t = row & (T_ - 1);
        out[((size_t)t * B_ + b) * O_ + o] = acc[j][v] + bias;
      }
    }
  }
}

// -------------------------------- launcher ---------------------------------

extern "C" void kernel_launch(void* const* d_in, const int* in_sizes, int n_in,
                              void* d_out, int out_size, void* d_ws, size_t ws_size,
                              hipStream_t stream) {
  (void)in_sizes; (void)n_in; (void)ws_size;
  const float* x    = (const float*)d_in[0];
  const float* spk  = (const float*)d_in[1];
  const float* wgw  = (const float*)d_in[2];
  const float* wgb  = (const float*)d_in[3];
  const float* Wih0 = (const float*)d_in[4];
  const float* Whh0 = (const float*)d_in[5];
  const float* bih0 = (const float*)d_in[6];
  const float* bhh0 = (const float*)d_in[7];
  const float* Wih1 = (const float*)d_in[8];
  const float* Whh1 = (const float*)d_in[9];
  const float* bih1 = (const float*)d_in[10];
  const float* bhh1 = (const float*)d_in[11];
  const float* fc1w = (const float*)d_in[12];
  const float* fc1b = (const float*)d_in[13];
  const float* fcw  = (const float*)d_in[14];
  const float* fcb  = (const float*)d_in[15];
  float* out = (float*)d_out;

  char* p = (char*)d_ws;
  auto carve = [&](size_t bytes) { char* r = p; p += (bytes + 255) & ~(size_t)255; return (void*)r; };
  bf16_t* xbf    = (bf16_t*)carve((size_t)B_ * T_ * D_ * 2);
  bf16_t* h0bf   = (bf16_t*)carve((size_t)E_ * B_ * T_ * F2_ * 2);
  bf16_t* h1bf   = (bf16_t*)carve((size_t)E_ * B_ * T_ * F2_ * 2);
  bf16_t* Wih0b  = (bf16_t*)carve((size_t)E_ * 2 * G4_ * D_ * 2);
  bf16_t* Whh0b  = (bf16_t*)carve((size_t)E_ * 2 * G4_ * H_ * 2);
  bf16_t* Wih1b  = (bf16_t*)carve((size_t)E_ * 2 * G4_ * F2_ * 2);
  bf16_t* Whh1b  = (bf16_t*)carve((size_t)E_ * 2 * G4_ * H_ * 2);
  bf16_t* fc1wb  = (bf16_t*)carve((size_t)H_ * F2_ * 2);
  bf16_t* fcwb   = (bf16_t*)carve((size_t)48 * H_ * 2);
  float*  bias0  = (float*)carve((size_t)E_ * 2 * G4_ * 4);
  float*  bias1  = (float*)carve((size_t)E_ * 2 * G4_ * 4);
  float*  gates  = (float*)carve((size_t)B_ * E_ * 4);
  bf16_t* dbf    = (bf16_t*)carve((size_t)B_ * T_ * H_ * 2);

  auto cvt = [&](const float* s, bf16_t* d, int n) {
    cvt_bf16_kernel<<<(n + 255) / 256, 256, 0, stream>>>(s, d, n);
  };
  cvt(x,    xbf,   B_ * T_ * D_);
  cvt(Wih0, Wih0b, E_ * 2 * G4_ * D_);
  cvt(Whh0, Whh0b, E_ * 2 * G4_ * H_);
  cvt(Wih1, Wih1b, E_ * 2 * G4_ * F2_);
  cvt(Whh1, Whh1b, E_ * 2 * G4_ * H_);
  cvt(fc1w, fc1wb, H_ * F2_);
  bias_sum_kernel<<<(E_ * 2 * G4_ + 255) / 256, 256, 0, stream>>>(bih0, bhh0, bias0, E_ * 2 * G4_);
  bias_sum_kernel<<<(E_ * 2 * G4_ + 255) / 256, 256, 0, stream>>>(bih1, bhh1, bias1, E_ * 2 * G4_);
  fcw_pad_kernel<<<(48 * H_ + 255) / 256, 256, 0, stream>>>(fcw, fcwb);

  gate_loss_kernel<<<1, 32, 0, stream>>>(spk, wgw, wgb, gates, out + (out_size - 1));

  // layer 0: input = shared x (expert stride 0), Fin = 128
  lstm_scan_kernel<<<E_ * 2, 256, 0, stream>>>(xbf, (size_t)0, D_, Wih0b, Whh0b, bias0, h0bf);
  // layer 1: input = per-expert h0, Fin = 256
  lstm_scan_kernel<<<E_ * 2, 256, 0, stream>>>(h0bf, (size_t)B_ * T_ * F2_, F2_, Wih1b, Whh1b, bias1, h1bf);

  fc1_gate_kernel<<<dim3((B_ * T_) / 128, 2), 256, 0, stream>>>(h1bf, fc1wb, fc1b, gates, dbf);
  fc_out_kernel<<<(B_ * T_) / 128, 256, 0, stream>>>(dbf, fcwb, fcb, out);
}
